// GraphNet_26585847562578
// MI455X (gfx1250) — matrix operand order repaired and checked by reading, not compiled, not run on hardware
//
#include <hip/hip_runtime.h>

// ---------------------------------------------------------------------------
// GraphSAGE + TopK pooling pipeline for MI455X (gfx1250, wave32, WMMA).
//
// Roofline reasoning:
//  * Edge scatter (SAGE2): 2M edges x 64 feats = 134M f32 atomics; the 25.6MB
//    accumulator lives in the 192MB L2 -> atomics resolve at L2, not HBM.
//  * SAGE2 GEMM: M=104,960, N=64, K=128 -> f16 WMMA (v_wmma_f32_16x16x32_f16)
//    with f32 accumulation. A fragments are loaded straight from global memory
//    as contiguous 32B runs (each lane's 16 contiguous K values form two 8-element
//    K-runs per lane); B fragments come from LDS with the weight matrix stored
//    TRANSPOSED so each fragment is one contiguous 32B LDS read (ds_load_b128).
//  * TopK: output is invariant to ordering inside the kept set, so rank-by-
//    counting per graph (O(N^2) in LDS) replaces a sort.
// ---------------------------------------------------------------------------

typedef __attribute__((ext_vector_type(16))) _Float16 v16h;
typedef __attribute__((ext_vector_type(8)))  float    v8f;

#define BB     256                   // graphs
#define NPG    512                   // nodes per graph
#define DEGC   16                    // edges per node
#define NE     (BB * NPG * DEGC)     // 2,097,152 edges
#define NNODE  (BB * NPG)            // 131,072 nodes
#define LB     5                     // lookback features
#define DIMC   64
#define OUTC   6
#define KP1    410                   // kept after pool1
#define KP2    328                   // kept after pool2
#define NP1    (BB * KP1)            // 104,960
#define NP2    (BB * KP2)            // 83,968

static_assert(NP1 % 128 == 0, "sage2 wmma grid must tile exactly");

// -------- 1/||w|| (tiny, single thread) ------------------------------------
__global__ void k_norm(const float* __restrict__ w, float* __restrict__ rn) {
  if (threadIdx.x == 0 && blockIdx.x == 0) {
    float s = 0.f;
    #pragma unroll
    for (int i = 0; i < DIMC; ++i) s += w[i] * w[i];
    rn[0] = rsqrtf(s);
  }
}

// -------- SAGE1 message scatter (x is [NN,5]) ------------------------------
__global__ void k_edge1(const int* __restrict__ src, const int* __restrict__ dst,
                        const float* __restrict__ x,
                        float* __restrict__ sum1, float* __restrict__ deg1) {
  int e = blockIdx.x * blockDim.x + threadIdx.x;
  if (e >= NE) return;
  int s = src[e], d = dst[e];
  atomicAdd(&deg1[d], 1.0f);
  #pragma unroll
  for (int f = 0; f < LB; ++f) atomicAdd(&sum1[d * LB + f], x[s * LB + f]);
}

// -------- SAGE1 combine: relu(mean@Wl + bl + x@Wr), K=5 (VALU) -------------
__global__ void k_sage1(const float* __restrict__ x, const float* __restrict__ sum1,
                        const float* __restrict__ deg1,
                        const float* __restrict__ Wl, const float* __restrict__ bl,
                        const float* __restrict__ Wr, float* __restrict__ h1) {
  int idx = blockIdx.x * blockDim.x + threadIdx.x;
  if (idx >= NNODE * DIMC) return;
  int i = idx >> 6, n = idx & 63;
  float invd = 1.0f / fmaxf(deg1[i], 1.0f);
  float acc = bl[n];
  #pragma unroll
  for (int k = 0; k < LB; ++k)
    acc += sum1[i * LB + k] * invd * Wl[k * DIMC + n] + x[i * LB + k] * Wr[k * DIMC + n];
  h1[idx] = fmaxf(acc, 0.0f);
}

// -------- pooling score: tanh(h.w / ||w||) ---------------------------------
__global__ void k_score(const float* __restrict__ h, const float* __restrict__ w,
                        const float* __restrict__ rn, float* __restrict__ score,
                        int nrows) {
  int i = blockIdx.x * blockDim.x + threadIdx.x;
  if (i >= nrows) return;
  float acc = 0.f;
  #pragma unroll 8
  for (int k = 0; k < DIMC; ++k) acc += h[i * DIMC + k] * w[k];
  score[i] = tanhf(acc * rn[0]);
}

// -------- top-k via rank counting (one block per graph) --------------------
__global__ void k_topk(const float* __restrict__ score, int npg, int k,
                       int* __restrict__ perm, float* __restrict__ vals,
                       int* __restrict__ inv) {
  __shared__ float ss[512];
  int g = blockIdx.x, t = threadIdx.x;
  float s = (t < npg) ? score[g * npg + t] : -1e30f;
  ss[t] = s;
  __syncthreads();
  if (t >= npg) return;
  int rank = 0;
  for (int j = 0; j < npg; ++j) {
    float sj = ss[j];
    rank += (sj > s) || (sj == s && j < t);   // stable, distinct ranks
  }
  if (rank < k) {
    int pos = g * k + rank;
    perm[pos] = g * npg + t;
    vals[pos] = s;
    if (inv) inv[g * npg + t] = pos;
  }
}

// -------- gather kept rows, scale by score ---------------------------------
__global__ void k_gather(const float* __restrict__ hin, const int* __restrict__ perm,
                         const float* __restrict__ vals, float* __restrict__ hout,
                         int nout) {
  int idx = blockIdx.x * blockDim.x + threadIdx.x;
  if (idx >= nout * DIMC) return;
  int p = idx >> 6, n = idx & 63;
  hout[idx] = hin[perm[p] * DIMC + n] * vals[p];
}

// -------- readout [gmp | gap] (one block per graph, 64 threads) ------------
__global__ void k_readout(const float* __restrict__ h, int k, float* __restrict__ xo) {
  int g = blockIdx.x, n = threadIdx.x;
  const float* base = h + (size_t)g * k * DIMC + n;
  float mx = -1e30f, sm = 0.f;
  for (int r = 0; r < k; ++r) { float v = base[(size_t)r * DIMC]; mx = fmaxf(mx, v); sm += v; }
  xo[g * 2 * DIMC + n]        = mx;
  xo[g * 2 * DIMC + DIMC + n] = sm / (float)k;
}

// -------- SAGE2 message scatter: 64 lanes per edge, atomics hit L2 ---------
__global__ void k_edge2(const int* __restrict__ src, const int* __restrict__ dst,
                        const int* __restrict__ inv, const float* __restrict__ hp1,
                        float* __restrict__ sum2, float* __restrict__ deg2) {
  int e = blockIdx.x * 4 + (threadIdx.x >> 6);
  int f = threadIdx.x & 63;
  if (e >= NE) return;
  int si = inv[src[e]], di = inv[dst[e]];
  if (si < 0 || di < 0) return;
  if (f == 0) atomicAdd(&deg2[di], 1.0f);
  atomicAdd(&sum2[di * DIMC + f], hp1[si * DIMC + f]);
}

// -------- SAGE2 combine as one WMMA GEMM -----------------------------------
// out = relu([mean | x] @ [Wl; Wr] + bl), M=NP1, N=64, K=128, f16 in, f32 acc.
// One wave per 16-row tile; 8 waves (256 thr) per block; grid = NP1/128 = 820.
//
// A fragments: contiguous K order (same as B) => lane(m,hi) holds K runs
//   [kb + hi*16, +8) and [kb + hi*16 + 8, +8): two contiguous 32B global loads,
//   scaled by 1/deg and converted to f16 in registers. Runs never straddle
//   the mean|x boundary (kb multiple of 32).
// B fragments: lane needs W[kb + hi*16 + j][col], j=0..15. Weights are staged
//   in LDS TRANSPOSED (sWt[n][k]) so each fragment is one contiguous 32B LDS
//   read -> ds_load_b128 pairs instead of 16 scalar ds_load_u16.
__global__ void __launch_bounds__(256) k_sage2_wmma(
    const float* __restrict__ sum2, const float* __restrict__ deg2,
    const float* __restrict__ hp1,  const float* __restrict__ Wl,
    const float* __restrict__ Wr,   const float* __restrict__ bl,
    float* __restrict__ h2) {
  __shared__ _Float16 sWt[64 * 128];     // transposed combined weights (16 KB)

  int tid  = threadIdx.x;
  int wave = tid >> 5;
  int lane = tid & 31;
  int m    = lane & 15;                  // A row / B,D col within tile
  int hi   = lane >> 4;

  // stage transposed weights once per block: sWt[n*128 + k] = Wcomb[k][n]
  for (int idx = tid; idx < 64 * 128; idx += 256) {
    int n = idx >> 7, k = idx & 127;
    float v = (k < 64) ? Wl[k * 64 + n] : Wr[(k - 64) * 64 + n];
    sWt[idx] = (_Float16)v;
  }
  __syncthreads();

  int tile = blockIdx.x * 8 + wave;
  int row  = tile * 16 + m;              // this lane's A row
  float invd = 1.0f / fmaxf(deg2[row], 1.0f);

  v8f acc[4] = {};                       // 4 N-tiles of 16 cols

  #pragma unroll
  for (int ks = 0; ks < 4; ++ks) {       // K = 128 in steps of 32
    const int kb = ks * 32;
    // ---- A fragment: two contiguous 8-float runs from global ----
    const float* abase = (kb < 64) ? (sum2 + (size_t)row * 64 + kb)
                                   : (hp1  + (size_t)row * 64 + (kb - 64));
    v8f r0 = *(const v8f*)(abase + hi * 16);       // K = kb + hi*16 ..+7
    v8f r1 = *(const v8f*)(abase + hi * 16 + 8);   // K = kb + hi*16 + 8 ..+7
    if (kb < 64) { r0 *= invd; r1 *= invd; }
    v16h a;
    #pragma unroll
    for (int j = 0; j < 8; ++j) {
      a[j]     = (_Float16)r0[j];
      a[j + 8] = (_Float16)r1[j];
    }
    // ---- 4 B fragments from transposed LDS weights, 4 WMMAs ----
    #pragma unroll
    for (int nt = 0; nt < 4; ++nt) {
      int col = nt * 16 + m;
      v16h b = *(const v16h*)(sWt + col * 128 + kb + hi * 16);  // 32B aligned
      acc[nt] = __builtin_amdgcn_wmma_f32_16x16x32_f16(
          false, a, false, b, (short)0, acc[nt], false, false);
    }
  }

  // D layout: VGPR r, lane -> M = r + 8*hi, N = m
  #pragma unroll
  for (int nt = 0; nt < 4; ++nt) {
    int col = nt * 16 + m;
    float bias = bl[col];
    #pragma unroll
    for (int r = 0; r < 8; ++r) {
      int grow = tile * 16 + r + (hi << 3);
      h2[grow * 64 + col] = fmaxf(acc[nt][r] + bias, 0.0f);
    }
  }
}

// -------- head: z2 = relu((x1+x2) @ lin1_W + b1) ---------------------------
__global__ void k_lin1(const float* __restrict__ x1, const float* __restrict__ x2,
                       const float* __restrict__ W, const float* __restrict__ bias,
                       float* __restrict__ z2) {
  int b = blockIdx.x, n = threadIdx.x;
  float acc = bias[n];
  #pragma unroll 8
  for (int k = 0; k < 128; ++k)
    acc += (x1[b * 128 + k] + x2[b * 128 + k]) * W[k * 64 + n];
  z2[b * 64 + n] = fmaxf(acc, 0.0f);
}

// -------- head: out = z2 @ lin2_W + b2 -------------------------------------
__global__ void k_lin2(const float* __restrict__ z2, const float* __restrict__ W,
                       const float* __restrict__ bias, float* __restrict__ out) {
  int b = blockIdx.x, o = threadIdx.x;
  if (o >= OUTC) return;
  float acc = bias[o];
  #pragma unroll 8
  for (int k = 0; k < 64; ++k) acc += z2[b * 64 + k] * W[k * OUTC + o];
  out[b * OUTC + o] = acc;
}

// ---------------------------------------------------------------------------
extern "C" void kernel_launch(void* const* d_in, const int* in_sizes, int n_in,
                              void* d_out, int out_size, void* d_ws, size_t ws_size,
                              hipStream_t stream) {
  (void)in_sizes; (void)n_in; (void)out_size; (void)ws_size;
  const float* x       = (const float*)d_in[0];
  const int*   ei      = (const int*)d_in[1];
  const int*   src     = ei;
  const int*   dst     = ei + NE;
  const float* c1_Wl   = (const float*)d_in[2];
  const float* c1_bl   = (const float*)d_in[3];
  const float* c1_Wr   = (const float*)d_in[4];
  const float* pool1_w = (const float*)d_in[5];
  const float* c2_Wl   = (const float*)d_in[6];
  const float* c2_bl   = (const float*)d_in[7];
  const float* c2_Wr   = (const float*)d_in[8];
  const float* pool2_w = (const float*)d_in[9];
  const float* lin1_W  = (const float*)d_in[10];
  const float* lin1_b  = (const float*)d_in[11];
  const float* lin2_W  = (const float*)d_in[12];
  const float* lin2_b  = (const float*)d_in[13];
  float* out = (float*)d_out;

  // ---- carve workspace (every buffer fully rewritten each launch) ----
  char* p = (char*)d_ws;
  auto carve = [&](size_t bytes) -> char* {
    char* r = p; p += (bytes + 255) & ~(size_t)255; return r;
  };
  float* rn     = (float*)carve(2 * sizeof(float));
  float* deg1   = (float*)carve((size_t)NNODE * sizeof(float));
  float* sum1   = (float*)carve((size_t)NNODE * LB * sizeof(float));
  float* h1     = (float*)carve((size_t)NNODE * DIMC * sizeof(float));
  float* score1 = (float*)carve((size_t)NNODE * sizeof(float));
  int*   perm1  = (int*)  carve((size_t)NP1 * sizeof(int));
  float* vals1  = (float*)carve((size_t)NP1 * sizeof(float));
  int*   inv    = (int*)  carve((size_t)NNODE * sizeof(int));
  float* hp1    = (float*)carve((size_t)NP1 * DIMC * sizeof(float));
  float* x1     = (float*)carve((size_t)BB * 2 * DIMC * sizeof(float));
  float* deg2   = (float*)carve((size_t)NP1 * sizeof(float));
  float* h2     = (float*)carve((size_t)NP1 * DIMC * sizeof(float));
  float* score2 = (float*)carve((size_t)NP1 * sizeof(float));
  int*   perm2  = (int*)  carve((size_t)NP2 * sizeof(int));
  float* vals2  = (float*)carve((size_t)NP2 * sizeof(float));
  float* x2     = (float*)carve((size_t)BB * 2 * DIMC * sizeof(float));
  float* z2     = (float*)carve((size_t)BB * DIMC * sizeof(float));
  // lifetime-disjoint aliases (ordering below guarantees no overlap in time):
  float* sum2 = h1;    // h1 dead after k_gather(pool1); sum2 zeroed after it
  float* hp2  = hp1;   // hp1 dead after k_sage2_wmma

  // ---- stage 0: norms + SAGE1 accumulators ----
  hipMemsetAsync(deg1, 0,    (size_t)NNODE * sizeof(float), stream);
  hipMemsetAsync(sum1, 0,    (size_t)NNODE * LB * sizeof(float), stream);
  hipMemsetAsync(inv,  0xFF, (size_t)NNODE * sizeof(int), stream);  // -1
  k_norm<<<1, 32, 0, stream>>>(pool1_w, rn);
  k_norm<<<1, 32, 0, stream>>>(pool2_w, rn + 1);

  // ---- SAGE1 + pool1 + readout1 ----
  k_edge1<<<NE / 256, 256, 0, stream>>>(src, dst, x, sum1, deg1);
  k_sage1<<<(NNODE * DIMC) / 256, 256, 0, stream>>>(x, sum1, deg1, c1_Wl, c1_bl, c1_Wr, h1);
  k_score<<<(NNODE + 255) / 256, 256, 0, stream>>>(h1, pool1_w, rn, score1, NNODE);
  k_topk<<<BB, 512, 0, stream>>>(score1, NPG, KP1, perm1, vals1, inv);
  k_gather<<<(NP1 * DIMC + 255) / 256, 256, 0, stream>>>(h1, perm1, vals1, hp1, NP1);
  k_readout<<<BB, DIMC, 0, stream>>>(hp1, KP1, x1);

  // ---- SAGE2 (WMMA) + pool2 + readout2 ----
  hipMemsetAsync(deg2, 0, (size_t)NP1 * sizeof(float), stream);
  hipMemsetAsync(sum2, 0, (size_t)NP1 * DIMC * sizeof(float), stream);
  k_edge2<<<NE / 4, 256, 0, stream>>>(src, dst, inv, hp1, sum2, deg2);
  k_sage2_wmma<<<NP1 / 128, 256, 0, stream>>>(sum2, deg2, hp1, c2_Wl, c2_Wr, c2_bl, h2);
  k_score<<<(NP1 + 255) / 256, 256, 0, stream>>>(h2, pool2_w, rn + 1, score2, NP1);
  k_topk<<<BB, 512, 0, stream>>>(score2, KP1, KP2, perm2, vals2, nullptr);
  k_gather<<<(NP2 * DIMC + 255) / 256, 256, 0, stream>>>(h2, perm2, vals2, hp2, NP2);
  k_readout<<<BB, DIMC, 0, stream>>>(hp2, KP2, x2);

  // ---- head MLP ----
  k_lin1<<<BB, DIMC, 0, stream>>>(x1, x2, lin1_W, lin1_b, z2);
  k_lin2<<<BB, 32, 0, stream>>>(z2, lin2_W, lin2_b, out);
}
